// SparseKAttention_71631464562857
// MI455X (gfx1250) — matrix operand order repaired and checked by reading, not compiled, not run on hardware
//
#include <hip/hip_runtime.h>
#include <hip/hip_bf16.h>

// ---------------------------------------------------------------------------
// CDNA5 (gfx1250) sparse-top-k attention pipeline.
// - all matmuls: v_wmma_f32_16x16x32_bf16 (wave32), bf16 operands, f32 accum
// - GEMM A panels staged LDS via Tensor Data Mover (tensor_load_to_lds +
//   s_wait_tensorcnt), B fragments are contiguous global b128 loads
// ---------------------------------------------------------------------------

typedef __attribute__((ext_vector_type(16))) __bf16   v16bf;
typedef __attribute__((ext_vector_type(8)))  __bf16   v8bf;
typedef __attribute__((ext_vector_type(8)))  float    v8f;
typedef __attribute__((ext_vector_type(4)))  unsigned u32x4;
typedef __attribute__((ext_vector_type(8)))  unsigned u32x8;

#define B_DIM  2
#define T_DIM  2048
#define D_DIM  1024
#define H_DIM  16
#define DH_DIM 64
#define BN_DIM 1024

__device__ __forceinline__ unsigned short bf16_bits(float f) {
  unsigned u = __builtin_bit_cast(unsigned, f);
  unsigned r = u + 0x7FFFu + ((u >> 16) & 1u);      // round-to-nearest-even
  return (unsigned short)(r >> 16);
}
__device__ __forceinline__ __bf16 f32_to_bf16(float f) {
  unsigned short h = bf16_bits(f);
  return __builtin_bit_cast(__bf16, h);
}

__device__ __forceinline__ v8f wmma_bf16(v16bf a, v16bf b, v8f c) {
  return __builtin_amdgcn_wmma_f32_16x16x32_bf16(false, a, false, b, (short)0, c,
                                                 false, false);
}

// ---- TDM: DMA a (16 x tile_w) bf16 tile from a row-major (tensor_h x
// tensor_w) tensor into LDS. 2D descriptor => 2-SGPR-group form (ISA §7/8).
__device__ __forceinline__ void tdm_load_16rows(const __bf16* gsrc,
                                                unsigned lds_off,
                                                unsigned tensor_w,
                                                unsigned tensor_h,
                                                unsigned tile_w) {
  unsigned long long ga = (unsigned long long)(size_t)gsrc;
  u32x4 g0;
  g0[0] = 1u;                                          // count=1 (user D#)
  g0[1] = lds_off;                                     // lds_addr (bytes)
  g0[2] = (unsigned)(ga & 0xFFFFFFFFull);              // global_addr[31:0]
  g0[3] = (unsigned)((ga >> 32) & 0x01FFFFFFull)       // global_addr[56:32]
          | (2u << 30);                                // type = 2 ("image")
  u32x8 g1;
  g1[0] = (1u << 16);                                  // data_size=1 (2B)
  g1[1] = (tensor_w & 0xFFFFu) << 16;                  // tensor_dim0[15:0]
  g1[2] = (tensor_w >> 16) | ((tensor_h & 0xFFFFu) << 16);  // dim0 hi|dim1 lo
  g1[3] = (tensor_h >> 16) | ((tile_w & 0xFFFFu) << 16);    // dim1 hi|tile_dim0
  g1[4] = 16u;                                         // tile_dim1=16
  g1[5] = tensor_w;                                    // dim0_stride[31:0]
  g1[6] = 0u;
  g1[7] = 0u;
  asm volatile("tensor_load_to_lds %0, %1" :: "s"(g0), "s"(g1) : "memory");
}

// ---- fragment builders (contiguous k = 16g + i order for both A and B, wave32) --------------
// A 16x32 bf16, row-major src: lane m=lane&15, g=lane>>4;
// element i -> k = 16g + i  => two contiguous 16B loads.
__device__ __forceinline__ v16bf a_frag_bf16(const __bf16* A, int lda, int k0) {
  const int lane = threadIdx.x & 31;
  const int m = lane & 15, g = lane >> 4;
  const __bf16* p = A + (size_t)m * lda + k0 + g * 16;
  v8bf lo = *(const v8bf*)(p);        // k = 16g .. 16g+7
  v8bf hi = *(const v8bf*)(p + 8);    // k = 16g+8 .. 16g+15
  return __builtin_shufflevector(lo, hi, 0, 1, 2, 3, 4, 5, 6, 7,
                                 8, 9, 10, 11, 12, 13, 14, 15);
}

// B 32x16 bf16 with B(k,n)=W[n][k] (W row-major [N x ldw]): lane n=lane&15,
// g=lane>>4; element i -> k = 16g + i  => one contiguous 32B load.
__device__ __forceinline__ v16bf b_frag_T_bf16(const __bf16* W, int ldw,
                                               int nbase, int k0) {
  const int lane = threadIdx.x & 31;
  const int n = lane & 15, g = lane >> 4;
  return *(const v16bf*)(W + (size_t)(nbase + n) * ldw + k0 + g * 16);
}

__device__ __forceinline__ int wave_reduce_add_i32(int v) {
#pragma unroll
  for (int o = 16; o > 0; o >>= 1) v += __shfl_xor(v, o, 32);
  return v;
}
__device__ __forceinline__ unsigned wave_reduce_max_u32(unsigned v) {
#pragma unroll
  for (int o = 16; o > 0; o >>= 1) {
    unsigned t = __shfl_xor(v, o, 32);
    v = (t > v) ? t : v;
  }
  return v;
}

// ---------------------------------------------------------------------------
// Kernel 0: f32 -> bf16 bulk convert (4 elements / thread)
// ---------------------------------------------------------------------------
__global__ void cvt_f32_to_bf16_kernel(const float* __restrict__ s,
                                       __bf16* __restrict__ d) {
  const size_t i = ((size_t)blockIdx.x * blockDim.x + threadIdx.x) * 4;
  float4 f = *(const float4*)(s + i);
  ushort4 o;
  o.x = bf16_bits(f.x);
  o.y = bf16_bits(f.y);
  o.z = bf16_bits(f.z);
  o.w = bf16_bits(f.w);
  *(ushort4*)((unsigned short*)d + i) = o;
}

// ---------------------------------------------------------------------------
// Kernel 1: QKV GEMM  C[4096x3072] = Xb[4096x1024] * Wb^T + bias.
// 128 threads = 4 waves; wave tile 16x64 (4 accumulators reuse one A frag).
// A panel (16x1024 bf16, 32KB) staged into LDS by the Tensor Data Mover.
// Q,K stored bf16 [B,H,T,Dh]; V stored bf16 TRANSPOSED [B,H,Dh,T].
// ---------------------------------------------------------------------------
__global__ void qkv_gemm_kernel(const __bf16* __restrict__ xb,
                                const __bf16* __restrict__ wb,
                                const float* __restrict__ bias,
                                __bf16* __restrict__ qw,
                                __bf16* __restrict__ kw,
                                __bf16* __restrict__ vw) {
  __shared__ __bf16 sA[16 * D_DIM];                  // 32 KB
  const int tid = threadIdx.x, lane = tid & 31, wv = tid >> 5;
  const int m0 = blockIdx.y * 16;                    // [0,4096)
  const int n0 = blockIdx.x * 256 + wv * 64;         // [0,3072)

  if (wv == 0) {
    tdm_load_16rows(xb + (size_t)m0 * D_DIM, (unsigned)(size_t)&sA[0],
                    D_DIM, 4096u, D_DIM);
    __builtin_amdgcn_s_wait_tensorcnt(0);
  }
  __syncthreads();

  v8f c0 = {}, c1 = {}, c2 = {}, c3 = {};
  for (int k0 = 0; k0 < D_DIM; k0 += 32) {
    __builtin_prefetch((const void*)(wb + (size_t)(n0 + (lane & 15)) * D_DIM +
                                     k0 + 128), 0, 1);
    v16bf a = a_frag_bf16((const __bf16*)sA, D_DIM, k0);
    c0 = wmma_bf16(a, b_frag_T_bf16(wb, D_DIM, n0, k0), c0);
    c1 = wmma_bf16(a, b_frag_T_bf16(wb, D_DIM, n0 + 16, k0), c1);
    c2 = wmma_bf16(a, b_frag_T_bf16(wb, D_DIM, n0 + 32, k0), c2);
    c3 = wmma_bf16(a, b_frag_T_bf16(wb, D_DIM, n0 + 48, k0), c3);
  }
  const int n = lane & 15, mg = lane >> 4;
  v8f cs[4] = {c0, c1, c2, c3};
#pragma unroll
  for (int j = 0; j < 4; ++j) {
#pragma unroll
    for (int r = 0; r < 8; ++r) {
      int m = r + mg * 8;
      int gm = m0 + m, gn = n0 + j * 16 + n;
      __bf16 vout = f32_to_bf16(cs[j][r] + bias[gn]);
      int bb = gm >> 11, t = gm & 2047;
      int s = gn >> 10, rem = gn & 1023, h = rem >> 6, dh = rem & 63;
      size_t bh = (size_t)bb * H_DIM + h;
      if (s == 0)      qw[(bh * T_DIM + t) * DH_DIM + dh] = vout;
      else if (s == 1) kw[(bh * T_DIM + t) * DH_DIM + dh] = vout;
      else             vw[(bh * DH_DIM + dh) * T_DIM + t] = vout;   // transposed
    }
  }
}

// ---------------------------------------------------------------------------
// Kernel 2: attention for one (b, h, 16-query tile). 256 threads = 8 waves.
// Dynamic LDS: sc 16x2048 f32 (128KB) + ps 16x2048 bf16 (64KB) + stats.
// partial-C buffer overlays sc (dead after stage 3). 192.4KB < 320KB/WGP.
// ---------------------------------------------------------------------------
__global__ void attn_kernel(const __bf16* __restrict__ qw,
                            const __bf16* __restrict__ kw,
                            const __bf16* __restrict__ vw,
                            const int* __restrict__ topk_p,
                            __bf16* __restrict__ y) {
  extern __shared__ char smem_raw[];
  float*  sc     = (float*)smem_raw;                          // 16*2048 f32
  __bf16* ps     = (__bf16*)(smem_raw + 16 * 2048 * 4);       // 16*2048 bf16
  float*  rowsum = (float*)(smem_raw + 16 * 2048 * 6);        // 16
  float*  rowmax = rowsum + 16;                               // 16
  float*  thr    = rowmax + 16;                               // 16
  float*  part   = sc;                                        // 8*16*16 overlay

  const int tid = threadIdx.x, lane = tid & 31, wv = tid >> 5;
  const int qt = blockIdx.x, h = blockIdx.y, b = blockIdx.z;
  const int t0 = qt * 16;
  const size_t bh = ((size_t)b * H_DIM + h);
  const __bf16* Q  = qw + (bh * T_DIM + t0) * DH_DIM;   // 16 x 64
  const __bf16* K  = kw + bh * T_DIM * DH_DIM;          // 2048 x 64
  const __bf16* Vt = vw + bh * DH_DIM * T_DIM;          // 64 x 2048 (Dh-major)
  const int topk = *topk_p;

  // ---- Stage 1: scores = (Q K^T) / sqrt(Dh) into LDS -------------------
  v16bf aq0 = a_frag_bf16(Q, DH_DIM, 0);
  v16bf aq1 = a_frag_bf16(Q, DH_DIM, 32);
  for (int kt = wv * 16; kt < wv * 16 + 16; ++kt) {
    const int n0 = kt * 16;
    v8f cacc = {};
    cacc = wmma_bf16(aq0, b_frag_T_bf16(K, DH_DIM, n0, 0), cacc);
    cacc = wmma_bf16(aq1, b_frag_T_bf16(K, DH_DIM, n0, 32), cacc);
    const int n = lane & 15, mg = lane >> 4;
#pragma unroll
    for (int r = 0; r < 8; ++r)
      sc[(r + mg * 8) * 2048 + n0 + n] = cacc[r] * 0.125f;
  }
  __syncthreads();

  // ---- Stage 2: exact top-k threshold + row max (radix select) ---------
  for (int rr = 0; rr < 2; ++rr) {
    const int row = wv * 2 + rr;
    unsigned uv[64];
    unsigned mx = 0u;
#pragma unroll
    for (int j = 0; j < 64; ++j) {
      float f = sc[row * 2048 + lane + j * 32];
      unsigned s = __builtin_bit_cast(unsigned, f);
      unsigned u = ((int)s < 0) ? ~s : (s | 0x80000000u);  // order-preserving
      uv[j] = u;
      mx = (u > mx) ? u : mx;
    }
    mx = wave_reduce_max_u32(mx);
    unsigned th = 0u;
    for (int bit = 31; bit >= 0; --bit) {
      unsigned cand = th | (1u << bit);
      int cnt = 0;
#pragma unroll
      for (int j = 0; j < 64; ++j) cnt += (uv[j] >= cand) ? 1 : 0;
      cnt = wave_reduce_add_i32(cnt);
      if (cnt >= topk) th = cand;  // max T with count(>=T) >= k == k-th largest
    }
    if (lane == 0) {
      unsigned s = (th & 0x80000000u) ? (th ^ 0x80000000u) : ~th;
      thr[row] = __builtin_bit_cast(float, s);
      unsigned sm = (mx & 0x80000000u) ? (mx ^ 0x80000000u) : ~mx;
      rowmax[row] = __builtin_bit_cast(float, sm);
    }
  }
  if (tid < 16) rowsum[tid] = 0.f;
  __syncthreads();

  // ---- Stage 3: p = (s>=thr) ? exp(s-max) : 0 -> ps (bf16), row sums ---
  {
    const int r = tid >> 4, seg = tid & 15;
    const float tv = thr[r], mxf = rowmax[r];
    float acc = 0.f;
    const int base = r * 2048 + seg * 128;
    for (int cci = 0; cci < 128; ++cci) {
      float s = sc[base + cci];
      float p = (s >= tv) ? __expf(s - mxf) : 0.f;
      ps[base + cci] = f32_to_bf16(p);
      acc += p;
    }
    atomicAdd(&rowsum[r], acc);
  }
  __syncthreads();   // sc dead from here; reused as `part`

  // ---- Stage 4: Y_tile = P @ V. wave = (ntile 0..3) x (k-half 0..1) ----
  {
    const int ntile = wv & 3, khalf = wv >> 2;
    const int n0 = ntile * 16, kb = khalf * 1024;
    v8f cacc = {};
    for (int k0 = kb; k0 < kb + 1024; k0 += 32) {
      v16bf a  = a_frag_bf16(ps, 2048, k0);            // P tile from LDS
      v16bf bb = b_frag_T_bf16(Vt, T_DIM, n0, k0);     // B(k,n)=Vt[n][k]
      cacc = wmma_bf16(a, bb, cacc);
    }
    const int n = lane & 15, mg = lane >> 4;
#pragma unroll
    for (int r = 0; r < 8; ++r)
      part[wv * 256 + (r + mg * 8) * 16 + n] = cacc[r];
  }
  __syncthreads();

  // combine K-halves, normalize, write y (bf16) in [B,T,BN] layout
  for (int idx = tid; idx < 16 * 64; idx += 256) {
    int m = idx >> 6, col = idx & 63;
    int w2 = col >> 4, n = col & 15;
    float vv = (part[w2 * 256 + m * 16 + n] + part[(w2 + 4) * 256 + m * 16 + n])
               / rowsum[m];
    y[((size_t)b * T_DIM + t0 + m) * BN_DIM + h * DH_DIM + col] = f32_to_bf16(vv);
  }
}

// ---------------------------------------------------------------------------
// Kernel 3: output projection  out[4096x1024] = Yb * Wprojb^T + bias (f32 out)
// A panel staged via TDM, same structure as kernel 1.
// ---------------------------------------------------------------------------
__global__ void proj_gemm_kernel(const __bf16* __restrict__ yb,
                                 const __bf16* __restrict__ wb,
                                 const float* __restrict__ bias,
                                 float* __restrict__ out) {
  __shared__ __bf16 sA[16 * BN_DIM];                 // 32 KB
  const int tid = threadIdx.x, lane = tid & 31, wv = tid >> 5;
  const int m0 = blockIdx.y * 16;
  const int n0 = blockIdx.x * 256 + wv * 64;

  if (wv == 0) {
    tdm_load_16rows(yb + (size_t)m0 * BN_DIM, (unsigned)(size_t)&sA[0],
                    BN_DIM, 4096u, BN_DIM);
    __builtin_amdgcn_s_wait_tensorcnt(0);
  }
  __syncthreads();

  v8f c0 = {}, c1 = {}, c2 = {}, c3 = {};
  for (int k0 = 0; k0 < BN_DIM; k0 += 32) {
    __builtin_prefetch((const void*)(wb + (size_t)(n0 + (lane & 15)) * BN_DIM +
                                     k0 + 128), 0, 1);
    v16bf a = a_frag_bf16((const __bf16*)sA, BN_DIM, k0);
    c0 = wmma_bf16(a, b_frag_T_bf16(wb, BN_DIM, n0, k0), c0);
    c1 = wmma_bf16(a, b_frag_T_bf16(wb, BN_DIM, n0 + 16, k0), c1);
    c2 = wmma_bf16(a, b_frag_T_bf16(wb, BN_DIM, n0 + 32, k0), c2);
    c3 = wmma_bf16(a, b_frag_T_bf16(wb, BN_DIM, n0 + 48, k0), c3);
  }
  const int n = lane & 15, mg = lane >> 4;
  v8f cs[4] = {c0, c1, c2, c3};
#pragma unroll
  for (int j = 0; j < 4; ++j) {
#pragma unroll
    for (int r = 0; r < 8; ++r) {
      int m = r + mg * 8, gn = n0 + j * 16 + n;
      out[(size_t)(m0 + m) * D_DIM + gn] = cs[j][r] + bias[gn];
    }
  }
}

// ---------------------------------------------------------------------------
extern "C" void kernel_launch(void* const* d_in, const int* in_sizes, int n_in,
                              void* d_out, int out_size, void* d_ws, size_t ws_size,
                              hipStream_t stream) {
  (void)in_sizes; (void)n_in; (void)out_size; (void)ws_size;
  const float* x      = (const float*)d_in[0];
  const float* w_qkv  = (const float*)d_in[1];
  const float* b_qkv  = (const float*)d_in[2];
  const float* w_proj = (const float*)d_in[3];
  const float* b_proj = (const float*)d_in[4];
  const int*   topk   = (const int*)d_in[5];
  float* out = (float*)d_out;

  // bf16 workspace layout (bytes):
  //   xb 8MB | wqkvb 6MB | wprojb 2MB | qw 8MB | kw 8MB | vw 8MB | yw 8MB = 48MB
  const size_t NX  = (size_t)B_DIM * T_DIM * D_DIM;        // 4,194,304
  const size_t NWQ = (size_t)3 * BN_DIM * D_DIM;           // 3,145,728
  const size_t NWP = (size_t)D_DIM * BN_DIM;               // 1,048,576
  const size_t NH  = (size_t)B_DIM * H_DIM * T_DIM * DH_DIM;

  char* w = (char*)d_ws;
  __bf16* xb     = (__bf16*)w;                 w += NX  * 2;
  __bf16* wqkvb  = (__bf16*)w;                 w += NWQ * 2;
  __bf16* wprojb = (__bf16*)w;                 w += NWP * 2;
  __bf16* qw     = (__bf16*)w;                 w += NH  * 2;
  __bf16* kw     = (__bf16*)w;                 w += NH  * 2;
  __bf16* vw     = (__bf16*)w;                 w += NH  * 2;
  __bf16* yw     = (__bf16*)w;

  // 0) one-time (per launch) down-converts
  cvt_f32_to_bf16_kernel<<<dim3(NX  / 1024), 256, 0, stream>>>(x, xb);
  cvt_f32_to_bf16_kernel<<<dim3(NWQ / 1024), 256, 0, stream>>>(w_qkv, wqkvb);
  cvt_f32_to_bf16_kernel<<<dim3(NWP / 1024), 256, 0, stream>>>(w_proj, wprojb);

  // 1) QKV projection: M=4096, N=3072 -> grid (3072/256, 4096/16)
  qkv_gemm_kernel<<<dim3(12, 256), 128, 0, stream>>>(xb, wqkvb, b_qkv,
                                                     qw, kw, vw);

  // 2) attention: (qtile, head, batch), 256 threads, 192.4KB dynamic LDS
  const size_t attn_lds = (size_t)16 * 2048 * 6 + 48 * 4;
  attn_kernel<<<dim3(T_DIM / 16, H_DIM, B_DIM), 256, attn_lds, stream>>>(
      qw, kw, vw, topk, yw);

  // 3) output projection: M=4096, N=1024 -> grid (1024/256, 4096/16)
  proj_gemm_kernel<<<dim3(4, 256), 128, 0, stream>>>(yw, wprojb, b_proj, out);
}
